// TransformerDuelingDQN_42082089566217
// MI455X (gfx1250) — matrix operand order repaired and checked by reading, compiled, not run on hardware
//
#include <hip/hip_runtime.h>
#include <math.h>

// ---------------- problem constants ----------------
#define BATCH 64
#define SEQ   256
#define DMODEL 1024
#define NHEAD 16
#define NLAYER 8
#define NADV  12
#define DENSE_IN 64
#define HEADDIM 64           // O
#define DFF   2048           // WF * D
#define MROWS (BATCH * SEQ)  // 16384
#define QKV_N (3 * DMODEL)   // 3072
#define ATT_SCALE 0.125f     // 64^-0.5

typedef __bf16 bf16;
typedef __attribute__((ext_vector_type(16))) __bf16 v16bf;
typedef __attribute__((ext_vector_type(8)))  float  v8f;
typedef __attribute__((ext_vector_type(4)))  unsigned int v4u;
typedef __attribute__((ext_vector_type(8)))  int  v8i;
typedef __attribute__((ext_vector_type(4)))  int  v4i;

#if defined(__AMDGCN__) && __has_builtin(__builtin_amdgcn_tensor_load_to_lds) && __has_builtin(__builtin_amdgcn_s_wait_tensorcnt)
#define HAVE_TDM 1
#endif

__device__ __forceinline__ bf16 f2bf(float f) {
  unsigned int u = __builtin_bit_cast(unsigned int, f);
  unsigned int r = u + 0x7FFFu + ((u >> 16) & 1u);   // round-to-nearest-even
  return __builtin_bit_cast(bf16, (unsigned short)(r >> 16));
}
__device__ __forceinline__ float gelu_exact(float x) {
  return 0.5f * x * (1.0f + erff(x * 0.70710678118654752f));
}

// ---------------- weight conversion ----------------
__global__ void k_f32_to_bf16(const float* __restrict__ in, bf16* __restrict__ out, size_t n) {
  size_t i = (size_t)blockIdx.x * blockDim.x + threadIdx.x;
  size_t stride = (size_t)gridDim.x * blockDim.x;
  for (; i < n; i += stride) out[i] = f2bf(in[i]);
}

// kqv_w (L, 3*H, D, O) -> per-layer row-major [D][3*H*O] (col j = ch*64 + o)
__global__ void k_cvt_kqv(const float* __restrict__ in, bf16* __restrict__ out) {
  size_t i = (size_t)blockIdx.x * blockDim.x + threadIdx.x;
  const size_t n = (size_t)NLAYER * 48 * DMODEL * HEADDIM;
  if (i >= n) return;
  int o  = (int)(i & 63);
  size_t t = i >> 6;
  int d  = (int)(t & (DMODEL - 1)); t >>= 10;
  int ch = (int)(t % 48);
  int l  = (int)(t / 48);
  out[(((size_t)l * DMODEL + d) * QKV_N) + ch * 64 + o] = f2bf(in[i]);
}

// ---------------- embed: dense token + concat + pos emb ----------------
__global__ __launch_bounds__(256)
void k_embed(const float* __restrict__ tokens, const float* __restrict__ side,
             const float* __restrict__ pe, const float* __restrict__ dw,
             const float* __restrict__ db, float* __restrict__ X, bf16* __restrict__ XB) {
  int row = blockIdx.x;           // b*256 + s
  int b = row >> 8, s = row & 255;
  for (int i = threadIdx.x; i < DMODEL; i += 256) {
    float v;
    if (s < SEQ - 1) {
      v = tokens[((size_t)b * (SEQ - 1) + s) * DMODEL + i];
    } else {
      v = db[i];
      #pragma unroll 8
      for (int k = 0; k < DENSE_IN; ++k) v += side[b * DENSE_IN + k] * dw[(size_t)k * DMODEL + i];
    }
    v += pe[(size_t)s * DMODEL + i];
    X [(size_t)row * DMODEL + i] = v;
    XB[(size_t)row * DMODEL + i] = f2bf(v);
  }
}

// ---------------- WMMA GEMM: C[M,N] = A[M,K]@B[K,N] (bf16 in, f32 acc) --------
// Block tile 128x128, 8 waves (4 over M x 2 over N), wave tile 32x64, K-step 64.
// A tile is DMA'd by the Tensor Data Mover when available; B tile is stored
// transposed in LDS so every fragment read is a contiguous 32B ds_load.
// EPI: 0 = bias -> bf16 out ; 1 = bias+gelu -> bf16 out ; 2 = bias -> f32 out
template <int EPI>
__global__ __launch_bounds__(256)
void k_gemm(const bf16* __restrict__ A, const bf16* __restrict__ Bw,
            const float* __restrict__ bias, void* __restrict__ Cout,
            int M, int N, int K) {
  __shared__ bf16 sA [128][72];   // row-major [m][k], 64 cols + 8 pad (144B pitch)
  __shared__ bf16 sBT[128][72];   // transposed [n][k]

  const int tid = threadIdx.x;
  const int wave = tid >> 5, lane = tid & 31;
  const int lrow = lane & 15, lhi = lane >> 4;
  const int wm = wave & 3;        // 4 waves over M (32 rows each)
  const int wn = wave >> 2;       // 2 waves over N (64 cols each)
  const int m0 = blockIdx.y * 128;
  const int n0 = blockIdx.x * 128;

  v8f acc[2][4];
  #pragma unroll
  for (int fm = 0; fm < 2; ++fm)
    #pragma unroll
    for (int fn = 0; fn < 4; ++fn) acc[fm][fn] = (v8f)(0.0f);

  for (int k0 = 0; k0 < K; k0 += 64) {
#if defined(HAVE_TDM)
    // ---- A tile 128x64 via Tensor Data Mover (one DMA per workgroup) ----
    if (wave == 0) {
      unsigned long long ga =
          (unsigned long long)(const void*)(A + (size_t)m0 * K + k0);
      v4u g0;
      g0[0] = 1u;                                   // count=1, user descriptor
      g0[1] = (unsigned)(size_t)&sA[0][0];          // lds_addr (LDS byte offset)
      g0[2] = (unsigned)(ga & 0xFFFFFFFFu);         // global_addr[31:0]
      g0[3] = (unsigned)((ga >> 32) & 0x01FFFFFFu)  // global_addr[56:32]
              | (2u << 30);                         // type = 2 (image)
      v8i g1;
      g1[0] = (int)((1u << 16)      // data_size = 1 -> 2 bytes
              | (1u << 20)          // pad_enable
              | (4u << 22)          // pad_interval: every 32 DWORDs (128B row)
              | (3u << 25));        // pad_amount: 4 DWORDs (16B -> 144B pitch)
      g1[1] = (int)(64u << 16);     // tensor_dim0 = 64 (lo16)
      g1[2] = (int)(128u << 16);    // tensor_dim0 hi=0 | tensor_dim1 = 128 (lo16)
      g1[3] = (int)(64u << 16);     // tensor_dim1 hi=0 | tile_dim0 = 64
      g1[4] = (int)128;             // tile_dim1 = 128, tile_dim2 = 0
      g1[5] = (int)K;               // tensor_dim0_stride[31:0] (data_size units)
      g1[6] = 0;                    // stride hi | tensor_dim1_stride lo
      g1[7] = 0;
      v4i z4 = (v4i)0;              // groups 2/3 unused (2D tensor)
      v8i z8 = (v8i)0;              // unused trailing group (6-arg toolchain form)
      __builtin_amdgcn_tensor_load_to_lds(g0, g1, z4, z4, z8, 0);
      __builtin_amdgcn_s_wait_tensorcnt(0);
    }
#else
    {  // fallback: A tile 128x64, 32 contiguous bf16 per thread
      int r = tid >> 1, seg = (tid & 1) * 32;
      const bf16* src = A + (size_t)(m0 + r) * K + k0 + seg;
      #pragma unroll
      for (int j = 0; j < 32; ++j) sA[r][seg + j] = src[j];
    }
#endif
    {  // B tile 64x128 -> transposed into sBT[n][k]
      int kk = tid >> 2, nseg = (tid & 3) * 32;
      const bf16* src = Bw + (size_t)(k0 + kk) * N + n0 + nseg;
      #pragma unroll
      for (int j = 0; j < 32; ++j) sBT[nseg + j][kk] = src[j];
    }
    if (k0 + 64 < K) {  // prefetch next tiles into L2 (global_prefetch_b8)
      __builtin_prefetch(A + (size_t)(m0 + (tid >> 1)) * K + k0 + 64 + (tid & 1) * 32, 0, 1);
      __builtin_prefetch(Bw + (size_t)(k0 + 64 + (tid >> 2)) * N + n0 + (tid & 3) * 32, 0, 1);
    }
    __syncthreads();

    #pragma unroll
    for (int kk = 0; kk < 2; ++kk) {          // two 16x16x32 K-slices per stage
      v16bf af[2];
      #pragma unroll
      for (int fm = 0; fm < 2; ++fm) {        // A frag: contiguous k (k = lhi*16 + j)
        int r = wm * 32 + fm * 16 + lrow;
        #pragma unroll
        for (int j = 0; j < 8; ++j) {
          af[fm][j]     = sA[r][kk * 32 + lhi * 16 + j];
          af[fm][8 + j] = sA[r][kk * 32 + lhi * 16 + 8 + j];
        }
      }
      #pragma unroll
      for (int fn = 0; fn < 4; ++fn) {        // B frag: contiguous from sBT
        v16bf bfg;
        int c = wn * 64 + fn * 16 + lrow;
        #pragma unroll
        for (int j = 0; j < 16; ++j) bfg[j] = sBT[c][kk * 32 + lhi * 16 + j];
        #pragma unroll
        for (int fm = 0; fm < 2; ++fm)
          acc[fm][fn] = __builtin_amdgcn_wmma_f32_16x16x32_bf16(
              false, af[fm], false, bfg, (short)0, acc[fm][fn], false, false);
      }
    }
    __syncthreads();
  }

  // epilogue (C layout: elem r -> row lhi*8+r, col lrow)
  #pragma unroll
  for (int fm = 0; fm < 2; ++fm)
    #pragma unroll
    for (int fn = 0; fn < 4; ++fn)
      #pragma unroll
      for (int r = 0; r < 8; ++r) {
        int grow = m0 + wm * 32 + fm * 16 + lhi * 8 + r;
        int gcol = n0 + wn * 64 + fn * 16 + lrow;
        float v = acc[fm][fn][r] + bias[gcol];
        if (EPI == 1) v = gelu_exact(v);
        if (EPI == 2) ((float*)Cout)[(size_t)grow * N + gcol] = v;
        else          ((bf16*)Cout)[(size_t)grow * N + gcol] = f2bf(v);
      }
}

// ---------------- flash attention (one WG per (b,h), 8 waves x 32 rows) ------
__global__ __launch_bounds__(256)
void k_attn(const bf16* __restrict__ qkv, float* __restrict__ Oout) {
  __shared__ bf16 sK[64][72];          // key block,  [t][o]
  __shared__ bf16 sVT[64][72];         // value block transposed, [o][t]
  __shared__ bf16 sP[8][32][72];       // per-wave P scratch

  const int h = blockIdx.x, b = blockIdx.y;
  const int tid = threadIdx.x, wave = tid >> 5, lane = tid & 31;
  const int lrow = lane & 15, lhi = lane >> 4;
  const int s0 = wave * 32;

  // Q fragments (A layout), K dim = 64 -> 2 k-frags, 32 rows -> 2 m-frags
  v16bf qf[2][2];
  #pragma unroll
  for (int fm = 0; fm < 2; ++fm)
    #pragma unroll
    for (int kf = 0; kf < 2; ++kf) {
      const bf16* qp = qkv + ((size_t)(b * SEQ + s0 + fm * 16 + lrow)) * QKV_N
                       + DMODEL + h * 64 + kf * 32;
      #pragma unroll
      for (int j = 0; j < 8; ++j) {
        qf[fm][kf][j]     = qp[lhi * 16 + j];
        qf[fm][kf][8 + j] = qp[lhi * 16 + 8 + j];
      }
    }

  float m_run[2][8], l_run[2][8];
  v8f acc_o[2][4];
  #pragma unroll
  for (int fm = 0; fm < 2; ++fm) {
    #pragma unroll
    for (int r = 0; r < 8; ++r) { m_run[fm][r] = -1e30f; l_run[fm][r] = 0.0f; }
    #pragma unroll
    for (int fn = 0; fn < 4; ++fn) acc_o[fm][fn] = (v8f)(0.0f);
  }

  for (int kb = 0; kb < 4; ++kb) {
    {  // cooperative load of K and V blocks (V stored transposed)
      int t = tid >> 2, seg = (tid & 3) * 16;
      const bf16* kp = qkv + ((size_t)(b * SEQ + kb * 64 + t)) * QKV_N + h * 64 + seg;
      const bf16* vp = qkv + ((size_t)(b * SEQ + kb * 64 + t)) * QKV_N + 2 * DMODEL + h * 64 + seg;
      #pragma unroll
      for (int j = 0; j < 16; ++j) sK[t][seg + j] = kp[j];
      #pragma unroll
      for (int j = 0; j < 16; ++j) sVT[seg + j][t] = vp[j];
    }
    __syncthreads();

    // S = Q @ K^T  (B frag element j -> k = kf*32 + lhi*16 + j, col = t')
    v8f s_acc[2][4];
    #pragma unroll
    for (int fm = 0; fm < 2; ++fm)
      #pragma unroll
      for (int fn = 0; fn < 4; ++fn) s_acc[fm][fn] = (v8f)(0.0f);
    #pragma unroll
    for (int kf = 0; kf < 2; ++kf)
      #pragma unroll
      for (int fn = 0; fn < 4; ++fn) {
        v16bf bfg;
        int c = fn * 16 + lrow;
        #pragma unroll
        for (int j = 0; j < 16; ++j) bfg[j] = sK[c][kf * 32 + lhi * 16 + j];
        #pragma unroll
        for (int fm = 0; fm < 2; ++fm)
          s_acc[fm][fn] = __builtin_amdgcn_wmma_f32_16x16x32_bf16(
              false, qf[fm][kf], false, bfg, (short)0, s_acc[fm][fn], false, false);
      }

    // online softmax per row (row lives across a 16-lane half at elem r)
    #pragma unroll
    for (int fm = 0; fm < 2; ++fm)
      #pragma unroll
      for (int r = 0; r < 8; ++r) {
        float mx = -1e30f;
        #pragma unroll
        for (int fn = 0; fn < 4; ++fn) {
          s_acc[fm][fn][r] *= ATT_SCALE;
          mx = fmaxf(mx, s_acc[fm][fn][r]);
        }
        #pragma unroll
        for (int d = 1; d < 16; d <<= 1) mx = fmaxf(mx, __shfl_xor(mx, d, 32));
        float mnew = fmaxf(m_run[fm][r], mx);
        float alpha = __expf(m_run[fm][r] - mnew);
        float rsum = 0.0f;
        #pragma unroll
        for (int fn = 0; fn < 4; ++fn) {
          float p = __expf(s_acc[fm][fn][r] - mnew);
          s_acc[fm][fn][r] = p;
          rsum += p;
        }
        #pragma unroll
        for (int d = 1; d < 16; d <<= 1) rsum += __shfl_xor(rsum, d, 32);
        l_run[fm][r] = l_run[fm][r] * alpha + rsum;
        m_run[fm][r] = mnew;
        #pragma unroll
        for (int fn = 0; fn < 4; ++fn) acc_o[fm][fn][r] *= alpha;
      }

    // C layout -> A layout via per-wave LDS bounce
    #pragma unroll
    for (int fm = 0; fm < 2; ++fm)
      #pragma unroll
      for (int fn = 0; fn < 4; ++fn)
        #pragma unroll
        for (int r = 0; r < 8; ++r)
          sP[wave][fm * 16 + lhi * 8 + r][fn * 16 + lrow] = f2bf(s_acc[fm][fn][r]);

    // O += P @ V
    #pragma unroll
    for (int kf = 0; kf < 2; ++kf) {
      v16bf pf[2];
      #pragma unroll
      for (int fm = 0; fm < 2; ++fm) {
        int rr = fm * 16 + lrow;
        #pragma unroll
        for (int j = 0; j < 8; ++j) {
          pf[fm][j]     = sP[wave][rr][kf * 32 + lhi * 16 + j];
          pf[fm][8 + j] = sP[wave][rr][kf * 32 + lhi * 16 + 8 + j];
        }
      }
      #pragma unroll
      for (int fn = 0; fn < 4; ++fn) {
        v16bf bfg;
        int c = fn * 16 + lrow;
        #pragma unroll
        for (int j = 0; j < 16; ++j) bfg[j] = sVT[c][kf * 32 + lhi * 16 + j];
        #pragma unroll
        for (int fm = 0; fm < 2; ++fm)
          acc_o[fm][fn] = __builtin_amdgcn_wmma_f32_16x16x32_bf16(
              false, pf[fm], false, bfg, (short)0, acc_o[fm][fn], false, false);
      }
    }
    __syncthreads();
  }

  // finalize: O /= l ; store to (b, s, h*64+o)
  #pragma unroll
  for (int fm = 0; fm < 2; ++fm)
    #pragma unroll
    for (int r = 0; r < 8; ++r) {
      float inv = 1.0f / l_run[fm][r];
      #pragma unroll
      for (int fn = 0; fn < 4; ++fn) {
        int s = s0 + fm * 16 + lhi * 8 + r;
        int col = h * 64 + fn * 16 + lrow;
        Oout[((size_t)(b * SEQ + s)) * DMODEL + col] = acc_o[fm][fn][r] * inv;
      }
    }
}

// ---------------- residual add + LayerNorm (one row per block) ---------------
__global__ __launch_bounds__(256)
void k_addln(const float* __restrict__ Ain, const float* __restrict__ Res,
             const float* __restrict__ g, const float* __restrict__ be,
             float* __restrict__ X, bf16* __restrict__ XB) {
  __shared__ float ssum[8], ssq[8], smean[2];
  int row = blockIdx.x, tid = threadIdx.x;
  const size_t base = (size_t)row * DMODEL;
  float v[4], s = 0.0f, sq = 0.0f;
  #pragma unroll
  for (int j = 0; j < 4; ++j) {
    int i = tid + j * 256;
    v[j] = Ain[base + i] + Res[base + i];
    s += v[j]; sq += v[j] * v[j];
  }
  #pragma unroll
  for (int d = 1; d < 32; d <<= 1) { s += __shfl_xor(s, d, 32); sq += __shfl_xor(sq, d, 32); }
  if ((tid & 31) == 0) { ssum[tid >> 5] = s; ssq[tid >> 5] = sq; }
  __syncthreads();
  if (tid == 0) {
    float ts = 0.0f, tq = 0.0f;
    #pragma unroll
    for (int w = 0; w < 8; ++w) { ts += ssum[w]; tq += ssq[w]; }
    float mean = ts * (1.0f / DMODEL);
    float var = tq * (1.0f / DMODEL) - mean * mean;
    smean[0] = mean; smean[1] = rsqrtf(var + 1e-5f);
  }
  __syncthreads();
  float mean = smean[0], rstd = smean[1];
  #pragma unroll
  for (int j = 0; j < 4; ++j) {
    int i = tid + j * 256;
    float y = (v[j] - mean) * rstd * g[i] + be[i];
    X [base + i] = y;
    XB[base + i] = f2bf(y);
  }
}

// ---------------- dueling heads ----------------
__global__ __launch_bounds__(256)
void k_head_mlp1(const float* __restrict__ X,
                 const float* __restrict__ w1v, const float* __restrict__ b1v,
                 const float* __restrict__ w1a, const float* __restrict__ b1a,
                 float* __restrict__ HV, float* __restrict__ HA) {
  int b = blockIdx.y;
  int j = blockIdx.x * 256 + threadIdx.x;      // 0..2047
  const float* last = X + ((size_t)b * SEQ + (SEQ - 1)) * DMODEL;
  float sv = b1v[j], sa = b1a[j];
  for (int d = 0; d < DMODEL; ++d) {
    float x = last[d];
    sv += x * w1v[(size_t)d * DFF + j];
    sa += x * w1a[(size_t)d * DFF + j];
  }
  HV[(size_t)b * DFF + j] = gelu_exact(sv);
  HA[(size_t)b * DFF + j] = gelu_exact(sa);
}

__global__ __launch_bounds__(256)
void k_head_out(const float* __restrict__ HV, const float* __restrict__ HA,
                const float* __restrict__ w2v, const float* __restrict__ b2v,
                const float* __restrict__ w2a, const float* __restrict__ b2a,
                float* __restrict__ out) {
  __shared__ float red[13][256];
  int b = blockIdx.x, tid = threadIdx.x;
  float pv = 0.0f, pa[NADV];
  #pragma unroll
  for (int a = 0; a < NADV; ++a) pa[a] = 0.0f;
  for (int j = tid; j < DFF; j += 256) {
    float hv = HV[(size_t)b * DFF + j];
    float ha = HA[(size_t)b * DFF + j];
    pv += hv * w2v[j];
    #pragma unroll
    for (int a = 0; a < NADV; ++a) pa[a] += ha * w2a[(size_t)j * NADV + a];
  }
  red[0][tid] = pv;
  #pragma unroll
  for (int a = 0; a < NADV; ++a) red[1 + a][tid] = pa[a];
  __syncthreads();
  for (int off = 128; off; off >>= 1) {
    if (tid < off)
      #pragma unroll
      for (int c = 0; c < 13; ++c) red[c][tid] += red[c][tid + off];
    __syncthreads();
  }
  if (tid == 0) {
    float val = red[0][0] + b2v[0];
    float adv[NADV], mean = 0.0f;
    #pragma unroll
    for (int a = 0; a < NADV; ++a) { adv[a] = red[1 + a][0] + b2a[a]; mean += adv[a]; }
    mean *= (1.0f / NADV);
    #pragma unroll
    for (int a = 0; a < NADV; ++a) out[b * NADV + a] = val + adv[a] - mean;
  }
}

// ---------------- host launcher ----------------
extern "C" void kernel_launch(void* const* d_in, const int* in_sizes, int n_in,
                              void* d_out, int out_size, void* d_ws, size_t ws_size,
                              hipStream_t stream) {
  const float* tokens  = (const float*)d_in[0];
  const float* side    = (const float*)d_in[1];
  const float* pos_emb = (const float*)d_in[2];
  const float* dense_w = (const float*)d_in[3];
  const float* dense_b = (const float*)d_in[4];
  const float* kqv_w   = (const float*)d_in[5];
  const float* kqv_b   = (const float*)d_in[6];
  const float* ffn_w1  = (const float*)d_in[7];
  const float* ffn_b1  = (const float*)d_in[8];
  const float* ffn_w2  = (const float*)d_in[9];
  const float* ffn_b2  = (const float*)d_in[10];
  const float* ln1_g   = (const float*)d_in[11];
  const float* ln1_b   = (const float*)d_in[12];
  const float* ln2_g   = (const float*)d_in[13];
  const float* ln2_b   = (const float*)d_in[14];
  const float* adv_w1  = (const float*)d_in[15];
  const float* adv_b1  = (const float*)d_in[16];
  const float* adv_w2  = (const float*)d_in[17];
  const float* adv_b2  = (const float*)d_in[18];
  const float* val_w1  = (const float*)d_in[19];
  const float* val_b1  = (const float*)d_in[20];
  const float* val_w2  = (const float*)d_in[21];
  const float* val_b2  = (const float*)d_in[22];
  (void)in_sizes; (void)n_in; (void)out_size; (void)ws_size;

  // workspace layout
  char* ws = (char*)d_ws;
  size_t off = 0;
  auto alloc = [&](size_t bytes) -> char* {
    char* p = ws + off;
    off = (off + bytes + 255) & ~(size_t)255;
    return p;
  };
  float* XF   = (float*)alloc((size_t)MROWS * DMODEL * 4);
  bf16*  XB   = (bf16*) alloc((size_t)MROWS * DMODEL * 2);
  bf16*  QKV  = (bf16*) alloc((size_t)MROWS * QKV_N * 2);
  float* OATT = (float*)alloc((size_t)MROWS * DMODEL * 4);
  bf16*  H1   = (bf16*) alloc((size_t)MROWS * DFF * 2);
  float* H2   = (float*)alloc((size_t)MROWS * DMODEL * 4);
  bf16*  WQKV = (bf16*) alloc((size_t)NLAYER * DMODEL * QKV_N * 2);
  bf16*  WF1  = (bf16*) alloc((size_t)NLAYER * DMODEL * DFF * 2);
  bf16*  WF2  = (bf16*) alloc((size_t)NLAYER * DFF * DMODEL * 2);
  float* HV   = (float*)alloc((size_t)BATCH * DFF * 4);
  float* HA   = (float*)alloc((size_t)BATCH * DFF * 4);

  // weight conversion
  {
    size_t n = (size_t)NLAYER * 48 * DMODEL * HEADDIM;
    k_cvt_kqv<<<dim3((unsigned)((n + 255) / 256)), 256, 0, stream>>>(kqv_w, WQKV);
    size_t n1 = (size_t)NLAYER * DMODEL * DFF;
    k_f32_to_bf16<<<4096, 256, 0, stream>>>(ffn_w1, WF1, n1);
    k_f32_to_bf16<<<4096, 256, 0, stream>>>(ffn_w2, WF2, n1);
  }

  // embed
  k_embed<<<MROWS, 256, 0, stream>>>(tokens, side, pos_emb, dense_w, dense_b, XF, XB);

  // layers
  for (int l = 0; l < NLAYER; ++l) {
    // QKV: (16384 x 1024) @ (1024 x 3072) + bias -> bf16
    k_gemm<0><<<dim3(QKV_N / 128, MROWS / 128), 256, 0, stream>>>(
        XB, WQKV + (size_t)l * DMODEL * QKV_N, kqv_b + (size_t)l * QKV_N,
        (void*)QKV, MROWS, QKV_N, DMODEL);
    // flash attention
    k_attn<<<dim3(NHEAD, BATCH), 256, 0, stream>>>(QKV, OATT);
    // x = LN(o + x)
    k_addln<<<MROWS, 256, 0, stream>>>(OATT, XF, ln1_g + (size_t)l * DMODEL,
                                       ln1_b + (size_t)l * DMODEL, XF, XB);
    // FFN1: + bias + gelu -> bf16
    k_gemm<1><<<dim3(DFF / 128, MROWS / 128), 256, 0, stream>>>(
        XB, WF1 + (size_t)l * DMODEL * DFF, ffn_b1 + (size_t)l * DFF,
        (void*)H1, MROWS, DFF, DMODEL);
    // FFN2: + bias -> f32
    k_gemm<2><<<dim3(DMODEL / 128, MROWS / 128), 256, 0, stream>>>(
        H1, WF2 + (size_t)l * DFF * DMODEL, ffn_b2 + (size_t)l * DMODEL,
        (void*)H2, MROWS, DMODEL, DFF);
    // x = LN(h + x)
    k_addln<<<MROWS, 256, 0, stream>>>(H2, XF, ln2_g + (size_t)l * DMODEL,
                                       ln2_b + (size_t)l * DMODEL, XF, XB);
  }

  // dueling heads
  k_head_mlp1<<<dim3(DFF / 256, BATCH), 256, 0, stream>>>(XF, val_w1, val_b1, adv_w1, adv_b1, HV, HA);
  k_head_out<<<BATCH, 256, 0, stream>>>(HV, HA, val_w2, val_b2, adv_w2, adv_b2, (float*)d_out);
}
